// SiglipAttention_10934986736073
// MI455X (gfx1250) — matrix operand order repaired and checked by reading, not compiled, not run on hardware
//
#include <hip/hip_runtime.h>

typedef __attribute__((ext_vector_type(16))) _Float16 v16h;
typedef __attribute__((ext_vector_type(8)))  _Float16 v8h;
typedef __attribute__((ext_vector_type(8)))  float    v8f;
typedef __attribute__((ext_vector_type(4)))  unsigned int v4u;
typedef __attribute__((ext_vector_type(8)))  int      v8i;
typedef __attribute__((ext_vector_type(4)))  int      v4i;

#define B_  32
#define S_  729
#define SP_ 736              // padded V row stride (32B-aligned rows, zero tail)
#define D_  1024
#define H_  16
#define HD_ 64
#define M_  (B_ * S_)        // 23328
#define SCALE_ 0.125f        // 64^-0.5

// TDM-tiled GEMM geometry
#define MT  128              // workgroup M tile
#define NT  128              // workgroup N tile
#define KB  64               // K slab per LDS buffer (128B per row), 2 WMMA k-substeps
#define LDSTRIDE 144         // bytes per LDS tile row: 128B data + 16B TDM pad (bank-conflict-free)
#define TILEBYTES (MT * LDSTRIDE)      // 18432
#define BUFSZ (2 * TILEBYTES)          // A + B tile       = 36864
#define SMEMSZ (2 * BUFSZ)             // double buffered  = 73728 (dynamic LDS)
#define MTILES ((M_ + MT - 1) / MT)    // 183
#define NTILES (D_ / NT)               // 8

// ---------------------------------------------------------------- LDS address helper
typedef __attribute__((address_space(3))) const char* lds_cptr_t;
__device__ __forceinline__ unsigned lds_addr_of(const void* p) {
  return (unsigned)(unsigned long long)(lds_cptr_t)p;
}

// ---------------------------------------------------------------- TDM descriptors
__device__ __forceinline__ v8i tdm_make_g1(unsigned rows_rem) {
  v8i g1;
  g1[0] = (int)((1u << 16)                 // data_size=1 (2 bytes)
              | (1u << 20)                 // pad_enable
              | (4u << 22)                 // pad_interval: 32 DW = 128B
              | (3u << 25));               // pad_amount:   4 DW = 16B
  g1[1] = (int)(((unsigned)D_ & 0xffffu) << 16);            // tensor_dim0[15:0]
  g1[2] = (int)((((unsigned)D_ >> 16) & 0xffffu)            // tensor_dim0[31:16]
              | ((rows_rem & 0xffffu) << 16));              // tensor_dim1[15:0]
  g1[3] = (int)(((rows_rem >> 16) & 0xffffu)                // tensor_dim1[31:16]
              | ((unsigned)KB << 16));                      // tile_dim0
  g1[4] = (int)((unsigned)MT & 0xffffu);                    // tile_dim1 (tile_dim2=0)
  g1[5] = (int)(unsigned)D_;                                // tensor_dim0_stride[31:0]
  g1[6] = 0;
  g1[7] = 0;
  return g1;
}

__device__ __forceinline__ void tdm_issue(unsigned lds_off, unsigned long long ga,
                                          const v8i& g1) {
  v4u g0;
  g0[0] = 1u;                                            // count=1 valid descriptor
  g0[1] = lds_off;                                       // lds_addr [63:32]
  g0[2] = (unsigned)(ga & 0xffffffffu);                  // global_addr [95:64]
  g0[3] = (unsigned)((ga >> 32) & 0x01ffffffu)           // global_addr [120:96]
        | (2u << 30);                                    // type=2 (image) [127:126]
  v4i gz = {0, 0, 0, 0};
#if __clang_major__ >= 23
  v8i gz8 = {0, 0, 0, 0, 0, 0, 0, 0};
  __builtin_amdgcn_tensor_load_to_lds(g0, g1, gz, gz, gz8, 0);
#else
  __builtin_amdgcn_tensor_load_to_lds(g0, g1, gz, gz, 0);
#endif
}

// ---------------------------------------------------------------- conversion / init
__global__ __launch_bounds__(256)
void f32_to_f16_kernel(const float* __restrict__ src, _Float16* __restrict__ dst, int n) {
  int i = blockIdx.x * 256 + threadIdx.x;
  if (i < n) dst[i] = (_Float16)src[i];
}

__global__ __launch_bounds__(256)
void zero_f16_kernel(_Float16* __restrict__ p, int n) {
  int i = blockIdx.x * 256 + threadIdx.x;
  if (i < n) p[i] = (_Float16)0.0f;
}

// ---------------------------------------------------------------- TDM-tiled projection GEMM
// C[m,n] = sum_k X[m,k]*W[n,k] + bias[n].
// Workgroup: 128x128 C tile, 8 waves as 4(M)x2(N); wave tile 32x64 = 2x4 WMMA frags.
// K loop: 64-wide slabs staged in LDS by TDM, double buffered, TENSORcnt-synced.
// mode 0: f16 out [B,H,S,HD]; mode 1: f16 out [B,H,HD,SP_] (V transposed, padded);
// mode 2: f32 out [M_,D_] (+bias) for the O projection.
__global__ __launch_bounds__(256)
void gemm_tdm_kernel(const _Float16* __restrict__ X,   // [M_, D_]
                     const _Float16* __restrict__ W,   // [D_, D_] (row=out,col=in)
                     const float*    __restrict__ bias,// [D_]
                     _Float16*       __restrict__ outH,
                     float*          __restrict__ outF,
                     int mode) {
  extern __shared__ __align__(16) char smem[];

  const int wave = threadIdx.x >> 5;
  const int lane = threadIdx.x & 31;
  const int mtile = blockIdx.x / NTILES;
  const int ntile = blockIdx.x - mtile * NTILES;
  const int m0 = mtile * MT;
  const int n0 = ntile * NT;
  const int wm = wave >> 1;          // 0..3 -> 32-row band
  const int wn = wave & 1;           // 0..1 -> 64-col band
  const int hi = lane >> 4;
  const int ml = lane & 15;

  const unsigned ldsBase = lds_addr_of(smem);
  const int NTK = D_ / KB;           // 16 K slabs

  const v8i g1A = tdm_make_g1((unsigned)(M_ - m0));
  const v8i g1B = tdm_make_g1((unsigned)(D_ - n0));
  const unsigned long long aBase = (unsigned long long)(X + (size_t)m0 * D_);
  const unsigned long long bBase = (unsigned long long)(W + (size_t)n0 * D_);

  v8f acc[2][4] = {};

  if (wave == 0) {
    tdm_issue(ldsBase, aBase, g1A);
    tdm_issue(ldsBase + TILEBYTES, bBase, g1B);
  }

  for (int kt = 0; kt < NTK; ++kt) {
    const int buf = kt & 1;
    if (wave == 0) {
      if (kt + 1 < NTK) {
        const unsigned long long koff = (unsigned long long)(kt + 1) * (KB * 2);
        const unsigned nb = ldsBase + (unsigned)((buf ^ 1) * BUFSZ);
        tdm_issue(nb, aBase + koff, g1A);
        tdm_issue(nb + TILEBYTES, bBase + koff, g1B);
        __builtin_amdgcn_s_wait_tensorcnt(2);  // current buffer's 2 tiles complete
      } else {
        __builtin_amdgcn_s_wait_tensorcnt(0);
      }
    }
    __syncthreads();                 // current buffer visible to all waves

    const char* Ab = smem + buf * BUFSZ;
    const char* Bb = Ab + TILEBYTES;

#pragma unroll
    for (int ks = 0; ks < 2; ++ks) { // two 32-wide k-substeps per slab
      const int kso = ks * 64;
      v16h af[2];
#pragma unroll
      for (int ms = 0; ms < 2; ++ms) {
        const char* rp = Ab + (wm * 32 + ms * 16 + ml) * LDSTRIDE + kso;
        v8h lo = *(const v8h*)(rp + 16 * hi);
        v8h hv = *(const v8h*)(rp + 32 + 16 * hi);
#pragma unroll
        for (int i = 0; i < 8; ++i) { af[ms][i] = lo[i]; af[ms][i + 8] = hv[i]; }
      }
      v16h bf[4];
#pragma unroll
      for (int j = 0; j < 4; ++j) {
        const char* rp = Bb + (wn * 64 + j * 16 + ml) * LDSTRIDE + kso;
        v8h lo = *(const v8h*)(rp + 16 * hi);
        v8h hv = *(const v8h*)(rp + 32 + 16 * hi);
#pragma unroll
        for (int i = 0; i < 8; ++i) { bf[j][i] = lo[i]; bf[j][i + 8] = hv[i]; }
      }
#pragma unroll
      for (int j = 0; j < 4; ++j)
#pragma unroll
        for (int ms = 0; ms < 2; ++ms)
          acc[ms][j] = __builtin_amdgcn_wmma_f32_16x16x32_f16(
              false, af[ms], false, bf[j], (short)0, acc[ms][j], false, false);
    }
    __syncthreads();                 // all waves done before buffer is overwritten
  }

#pragma unroll
  for (int ms = 0; ms < 2; ++ms) {
#pragma unroll
    for (int j = 0; j < 4; ++j) {
      const int col = n0 + wn * 64 + j * 16 + ml;
      const float bv = bias[col];
      const int h = col >> 6, hd = col & 63;
#pragma unroll
      for (int r = 0; r < 8; ++r) {
        const int mm = m0 + wm * 32 + ms * 16 + r + 8 * hi;
        if (mm >= M_) continue;
        const float v = acc[ms][j][r] + bv;
        if (mode == 2) {
          outF[(size_t)mm * D_ + col] = v;
        } else {
          const int bb = mm / S_;
          const int ss = mm - bb * S_;
          const size_t idx = (mode == 1)
              ? (((size_t)(bb * H_ + h) * HD_ + hd) * SP_ + ss)
              : (((size_t)(bb * H_ + h) * S_ + ss) * HD_ + hd);
          outH[idx] = (_Float16)v;
        }
      }
    }
  }
}

// ---------------------------------------------------------------- fused scores + softmax
// One workgroup per (bh, 16-row band): 8 waves x 6 column tiles cover all 729 cols
// (12 WMMAs/wave). Scores stay in registers; row max/sum via ml-shuffles + 16x8 LDS
// cross-wave reduction; only final probabilities are written (1 pass over attn).
__global__ __launch_bounds__(256)
void scores_softmax_kernel(const _Float16* __restrict__ Q,   // [B,H,S,HD]
                           const _Float16* __restrict__ Km,  // [B,H,S,HD]
                           float* __restrict__ attn) {       // [B,H,S,S]
  __shared__ float lds_red[16][8];
  const int wave = threadIdx.x >> 5;
  const int lane = threadIdx.x & 31;
  const int bh    = blockIdx.x / 46;
  const int mtile = blockIdx.x - bh * 46;
  const int hi = lane >> 4;
  const int ml = lane & 15;

  const int m  = mtile * 16 + ml;
  const int mc = (m < S_) ? m : (S_ - 1);
  const _Float16* qrow = Q + ((size_t)bh * S_ + mc) * HD_;

  v16h af[2];
#pragma unroll
  for (int ks = 0; ks < 2; ++ks) {
    const int k0 = ks * 32;
    v8h lo = *(const v8h*)(qrow + k0 + 16 * hi);
    v8h hv = *(const v8h*)(qrow + k0 + 16 * hi + 8);
#pragma unroll
    for (int i = 0; i < 8; ++i) { af[ks][i] = lo[i]; af[ks][i + 8] = hv[i]; }
  }

  v8f acc[6] = {};
#pragma unroll
  for (int jj = 0; jj < 6; ++jj) {
    const int n  = wave * 96 + jj * 16 + ml;
    const int nc = (n < S_) ? n : (S_ - 1);
    const _Float16* krow = Km + ((size_t)bh * S_ + nc) * HD_;
#pragma unroll
    for (int ks = 0; ks < 2; ++ks) {
      v16h b = *(const v16h*)(krow + ks * 32 + 16 * hi);
      acc[jj] = __builtin_amdgcn_wmma_f32_16x16x32_f16(
          false, af[ks], false, b, (short)0, acc[jj], false, false);
    }
  }

  // scale + mask invalid columns
  float s[6][8];
#pragma unroll
  for (int jj = 0; jj < 6; ++jj) {
    const bool valid = (wave * 96 + jj * 16 + ml) < S_;
#pragma unroll
    for (int r = 0; r < 8; ++r)
      s[jj][r] = valid ? acc[jj][r] * SCALE_ : -3.0e38f;
  }

  // row max: in-register over jj, shuffle over ml, LDS over waves
  float pm[8];
#pragma unroll
  for (int r = 0; r < 8; ++r) {
    float v = s[0][r];
#pragma unroll
    for (int jj = 1; jj < 6; ++jj) v = fmaxf(v, s[jj][r]);
    pm[r] = v;
  }
#pragma unroll
  for (int off = 1; off < 16; off <<= 1)
#pragma unroll
    for (int r = 0; r < 8; ++r) pm[r] = fmaxf(pm[r], __shfl_xor(pm[r], off, 32));
  if (ml == 0)
#pragma unroll
    for (int r = 0; r < 8; ++r) lds_red[r + 8 * hi][wave] = pm[r];
  __syncthreads();
  float rm[8];
#pragma unroll
  for (int r = 0; r < 8; ++r) {
    float v = lds_red[r + 8 * hi][0];
#pragma unroll
    for (int w2 = 1; w2 < 8; ++w2) v = fmaxf(v, lds_red[r + 8 * hi][w2]);
    rm[r] = v;
  }
  __syncthreads();                   // reuse lds_red for sums

  // exp + row sum
  float ps[8];
#pragma unroll
  for (int r = 0; r < 8; ++r) ps[r] = 0.f;
#pragma unroll
  for (int jj = 0; jj < 6; ++jj)
#pragma unroll
    for (int r = 0; r < 8; ++r) {
      const float e = __expf(s[jj][r] - rm[r]);  // masked cols underflow to 0
      s[jj][r] = e;
      ps[r] += e;
    }
#pragma unroll
  for (int off = 1; off < 16; off <<= 1)
#pragma unroll
    for (int r = 0; r < 8; ++r) ps[r] += __shfl_xor(ps[r], off, 32);
  if (ml == 0)
#pragma unroll
    for (int r = 0; r < 8; ++r) lds_red[r + 8 * hi][wave] = ps[r];
  __syncthreads();
#pragma unroll
  for (int r = 0; r < 8; ++r) {
    float v = 0.f;
#pragma unroll
    for (int w2 = 0; w2 < 8; ++w2) v += lds_red[r + 8 * hi][w2];
    rm[r] = 1.0f / v;                // reuse rm as inv-sum
  }

  // write probabilities
#pragma unroll
  for (int jj = 0; jj < 6; ++jj) {
    const int n = wave * 96 + jj * 16 + ml;
    if (n >= S_) continue;
#pragma unroll
    for (int r = 0; r < 8; ++r) {
      const int mm = mtile * 16 + r + 8 * hi;
      if (mm < S_)
        attn[((size_t)bh * S_ + mm) * S_ + n] = s[jj][r] * rm[r];
    }
  }
}

// ---------------------------------------------------------------- ctx = attn @ V
// A = attn rows (f32 -> f16 on the fly): 22 unguarded K-steps + 1 guarded tail.
// B = Vt rows with padded stride SP_=736 (zeroed tail) -> clean 32B vector loads.
__global__ __launch_bounds__(256)
void ctx_kernel(const float* __restrict__ attn,     // [B,H,S,S]
                const _Float16* __restrict__ Vt,    // [B,H,HD,SP_]
                _Float16* __restrict__ ctx) {       // [M_, D_]
  const int wave = threadIdx.x >> 5;
  const int lane = threadIdx.x & 31;
  const int gid  = blockIdx.x * 8 + wave;      // 512*46 strips (exact)
  const int bh    = gid / 46;
  const int mtile = gid - bh * 46;
  const int hi = lane >> 4;
  const int ml = lane & 15;

  const int m  = mtile * 16 + ml;
  const int mc = (m < S_) ? m : (S_ - 1);
  const float* arow = attn + ((size_t)bh * S_ + mc) * S_;
  const _Float16* vbase = Vt + (size_t)bh * HD_ * SP_;
  v8f acc[4] = {};

  for (int k0 = 0; k0 < 704; k0 += 32) {       // all k <= 703+32 < 736, valid < 729
    v16h a;
    const float* p0 = arow + k0 + 16 * hi;
    const float* p1 = p0 + 8;
#pragma unroll
    for (int i = 0; i < 8; ++i) { a[i] = (_Float16)p0[i]; a[i + 8] = (_Float16)p1[i]; }
#pragma unroll
    for (int j = 0; j < 4; ++j) {
      v16h b = *(const v16h*)(vbase + (size_t)(j * 16 + ml) * SP_ + k0 + 16 * hi);
      acc[j] = __builtin_amdgcn_wmma_f32_16x16x32_f16(
          false, a, false, b, (short)0, acc[j], false, false);
    }
  }
  {                                            // tail k0=704: guard attn, V pad is zero
    const int k0 = 704;
    v16h a;
#pragma unroll
    for (int e = 0; e < 16; ++e) {
      const int k = k0 + 16 * hi + e;
      a[e] = (k < S_) ? (_Float16)arow[k] : (_Float16)0.0f;
    }
#pragma unroll
    for (int j = 0; j < 4; ++j) {
      v16h b = *(const v16h*)(vbase + (size_t)(j * 16 + ml) * SP_ + k0 + 16 * hi);
      acc[j] = __builtin_amdgcn_wmma_f32_16x16x32_f16(
          false, a, false, b, (short)0, acc[j], false, false);
    }
  }

  const int h  = bh & (H_ - 1);
  const int bb = bh >> 4;
#pragma unroll
  for (int j = 0; j < 4; ++j) {
#pragma unroll
    for (int r = 0; r < 8; ++r) {
      const int mm = mtile * 16 + r + 8 * hi;
      if (mm < S_)
        ctx[((size_t)bb * S_ + mm) * D_ + h * HD_ + j * 16 + ml] = (_Float16)acc[j][r];
    }
  }
}

// ---------------------------------------------------------------- launch
extern "C" void kernel_launch(void* const* d_in, const int* in_sizes, int n_in,
                              void* d_out, int out_size, void* d_ws, size_t ws_size,
                              hipStream_t stream) {
  const float* hs = (const float*)d_in[0];
  const float* Wq = (const float*)d_in[1];
  const float* bq = (const float*)d_in[2];
  const float* Wk = (const float*)d_in[3];
  const float* bk = (const float*)d_in[4];
  const float* Wv = (const float*)d_in[5];
  const float* bv = (const float*)d_in[6];
  const float* Wo = (const float*)d_in[7];
  const float* bo = (const float*)d_in[8];

  float* out  = (float*)d_out;                 // [B,S,D]
  float* attn = out + (size_t)M_ * D_;         // [B,H,S,S]

  char* ws = (char*)d_ws;
  size_t off = 0;
  auto alloc = [&](size_t bytes) -> void* {
    void* p = ws + off;
    off = (off + bytes + 255) & ~(size_t)255;
    return p;
  };
  _Float16* Xh  = (_Float16*)alloc((size_t)M_ * D_ * sizeof(_Float16));
  _Float16* Wqh = (_Float16*)alloc((size_t)D_ * D_ * sizeof(_Float16));
  _Float16* Wkh = (_Float16*)alloc((size_t)D_ * D_ * sizeof(_Float16));
  _Float16* Wvh = (_Float16*)alloc((size_t)D_ * D_ * sizeof(_Float16));
  _Float16* Woh = (_Float16*)alloc((size_t)D_ * D_ * sizeof(_Float16));
  _Float16* Qh  = (_Float16*)alloc((size_t)M_ * D_ * sizeof(_Float16));       // [B,H,S,HD]
  _Float16* Kh  = (_Float16*)alloc((size_t)M_ * D_ * sizeof(_Float16));       // [B,H,S,HD]
  _Float16* Vth = (_Float16*)alloc((size_t)B_ * H_ * HD_ * SP_ * sizeof(_Float16)); // [B,H,HD,SP_]
  _Float16* Ch  = (_Float16*)alloc((size_t)M_ * D_ * sizeof(_Float16));       // [M_, D_]

  const int nX = M_ * D_;
  const int nW = D_ * D_;
  const int nV = B_ * H_ * HD_ * SP_;
  f32_to_f16_kernel<<<(nX + 255) / 256, 256, 0, stream>>>(hs, Xh, nX);
  f32_to_f16_kernel<<<(nW + 255) / 256, 256, 0, stream>>>(Wq, Wqh, nW);
  f32_to_f16_kernel<<<(nW + 255) / 256, 256, 0, stream>>>(Wk, Wkh, nW);
  f32_to_f16_kernel<<<(nW + 255) / 256, 256, 0, stream>>>(Wv, Wvh, nW);
  f32_to_f16_kernel<<<(nW + 255) / 256, 256, 0, stream>>>(Wo, Woh, nW);
  zero_f16_kernel<<<(nV + 255) / 256, 256, 0, stream>>>(Vth, nV);

  const int gemmBlocks = MTILES * NTILES;             // 183*8 = 1464
  gemm_tdm_kernel<<<gemmBlocks, 256, SMEMSZ, stream>>>(Xh, Wqh, bq, Qh,  nullptr, 0);
  gemm_tdm_kernel<<<gemmBlocks, 256, SMEMSZ, stream>>>(Xh, Wkh, bk, Kh,  nullptr, 0);
  gemm_tdm_kernel<<<gemmBlocks, 256, SMEMSZ, stream>>>(Xh, Wvh, bv, Vth, nullptr, 1);

  scores_softmax_kernel<<<B_ * H_ * 46, 256, 0, stream>>>(Qh, Kh, attn);  // 23552 blocks

  const int ctxBlocks = (B_ * H_ * 46) / 8;           // 2944
  ctx_kernel<<<ctxBlocks, 256, 0, stream>>>(attn, Vth, Ch);

  gemm_tdm_kernel<<<gemmBlocks, 256, SMEMSZ, stream>>>(Ch, Woh, bo, nullptr, out, 2);
}
